// SLAHead_28303834481290
// MI455X (gfx1250) — compile-verified, matrix operand order checked by reading, not run on hardware
//
#include <hip/hip_runtime.h>
#include <hip/hip_bf16.h>

typedef __attribute__((ext_vector_type(16))) _Float16 v16h;
typedef __attribute__((ext_vector_type(8)))  _Float16 v8h;
typedef __attribute__((ext_vector_type(8)))  float    v8f;

#define B_    64
#define C_    512
#define T_    784
#define H_    512
#define NC_   30
#define LOC_  8
#define STEPS 501
#define XK    544      // 512 + 30 padded to multiple of 32
#define GN    1536     // 3*H

// ---------------- fast device math ----------------

__device__ __forceinline__ float fast_tanh(float x) {
#if __has_builtin(__builtin_amdgcn_tanhf)
  return __builtin_amdgcn_tanhf(x);        // native V_TANH_F32 (gfx1250 TRANS op)
#else
  return tanhf(x);
#endif
}

__device__ __forceinline__ float sigm(float x) { return 1.0f / (1.0f + __expf(-x)); }

// ---------------- WMMA helpers (CDNA5 16x16x32 f16 -> f32) ----------------

__device__ __forceinline__ v8f wmma16(v16h a, v16h b, v8f c) {
  // 8 args: (neg_a, A, neg_b, B, c_mod, C, reuse_a, reuse_b)
  return __builtin_amdgcn_wmma_f32_16x16x32_f16(false, a, false, b, (short)0, c,
                                                false, false);
}

// packed B tile: [(nt*ktiles+kt)*32 + lane]*16 halves, in ISA lane/VGPR order
__device__ __forceinline__ v16h load_b_tile(const _Float16* pk, int ktiles,
                                            int nt, int kt, int lane) {
  return *(const v16h*)(pk + (((size_t)nt * ktiles + kt) * 32 + lane) * 16);
}

// A tile from f32 row-major activation [M, ld]; ISA 16-bit A layout:
// lane l: m = l&15, hi = l>>4; halves 0..7 -> K = hi*8+0..7, 8..15 -> K = 16+hi*8+0..7
__device__ __forceinline__ v16h load_a_f32(const float* A, int ld, int row,
                                           int kt, int lane) {
  const float* p = A + (size_t)(row + (lane & 15)) * ld + kt * 32 + (lane >> 4) * 8;
  v16h a;
#pragma unroll
  for (int i = 0; i < 8; ++i) a[i] = (_Float16)p[i];
#pragma unroll
  for (int i = 0; i < 8; ++i) a[8 + i] = (_Float16)p[16 + i];
  return a;
}

__device__ __forceinline__ v16h load_a_f16(const _Float16* A, int ld, int row,
                                           int kt, int lane) {
  const _Float16* p = A + (size_t)(row + (lane & 15)) * ld + kt * 32 + (lane >> 4) * 8;
  v8h lo = *(const v8h*)p;
  v8h hi = *(const v8h*)(p + 16);
  v16h a;
#pragma unroll
  for (int i = 0; i < 8; ++i) { a[i] = lo[i]; a[8 + i] = hi[i]; }
  return a;
}

// ---------------- Phase-0 kernels ----------------

// pack f32 weight W[N,K] into WMMA-B f16 tiles (Np x Kp padded, zero-filled)
__global__ void pack_weight(const float* __restrict__ W, _Float16* __restrict__ out,
                            int N, int K, int Np, int Kp) {
  int idx = blockIdx.x * blockDim.x + threadIdx.x;
  int ktiles = Kp >> 5;
  int total = (Np >> 4) * ktiles * 512;
  if (idx >= total) return;
  int j    = idx & 15;
  int lane = (idx >> 4) & 31;
  int tile = idx >> 9;
  int kt = tile % ktiles;
  int nt = tile / ktiles;
  int n  = nt * 16 + (lane & 15);
  int hi = lane >> 4;
  int k  = kt * 32 + (j < 8 ? hi * 8 + j : 16 + hi * 8 + (j - 8));
  float v = (n < N && k < K) ? W[(size_t)n * K + k] : 0.0f;
  out[idx] = (_Float16)v;
}

__global__ void cvt_f32_f16(const float* __restrict__ in, _Float16* __restrict__ out, int n) {
  int i = blockIdx.x * blockDim.x + threadIdx.x;
  if (i < n) out[i] = (_Float16)in[i];
}

// fea [B,C,T] f32 -> featT [B,T,C] f16
__global__ void transpose_feat(const float* __restrict__ fea, _Float16* __restrict__ featT) {
  size_t idx = (size_t)blockIdx.x * blockDim.x + threadIdx.x;
  if (idx >= (size_t)B_ * T_ * C_) return;
  int c = (int)(idx % C_);
  int t = (int)((idx / C_) % T_);
  int b = (int)(idx / ((size_t)C_ * T_));
  featT[idx] = (_Float16)fea[((size_t)b * C_ + c) * T_ + t];
}

__global__ void init_state(float* hid, int* chars) {
  int i = blockIdx.x * blockDim.x + threadIdx.x;
  if (i < B_ * H_) hid[i] = 0.0f;
  if (i < B_) chars[i] = 0;
}

// feat_proj[b,t,h] = sum_c featT[b,t,c] * i2h[h,c]  (f16 out, WMMA)
__global__ void __launch_bounds__(256)
featproj_gemm(const _Float16* __restrict__ featT, const _Float16* __restrict__ pkI2h,
              _Float16* __restrict__ fp) {
  int wave = (blockIdx.x * blockDim.x + threadIdx.x) >> 5;  // 64*49*32 waves
  int lane = threadIdx.x & 31;
  int nt = wave & 31;
  int rem = wave >> 5;
  int mt = rem % 49;
  int b  = rem / 49;
  if (b >= B_) return;  // wave-uniform
  const _Float16* A = featT + (size_t)b * T_ * C_;
  v8f acc = {};
#pragma unroll 4
  for (int kt = 0; kt < 16; ++kt) {
    v16h a  = load_a_f16(A, C_, mt * 16, kt, lane);
    v16h bb = load_b_tile(pkI2h, 16, nt, kt, lane);
    acc = wmma16(a, bb, acc);
  }
  int n  = nt * 16 + (lane & 15);
  int m0 = mt * 16 + (lane >> 4) * 8;
  _Float16* D = fp + (size_t)b * T_ * H_;
#pragma unroll
  for (int r = 0; r < 8; ++r) D[(size_t)(m0 + r) * H_ + n] = (_Float16)acc[r];
}

// ---------------- Per-step kernel 1: attention + context ----------------
// one block per batch row b, 512 threads (16 waves)
__global__ void __launch_bounds__(512)
attn_step(const _Float16* __restrict__ featT, const _Float16* __restrict__ fproj,
          const _Float16* __restrict__ h2h16, const float* __restrict__ h2h_b,
          const float* __restrict__ sv, const float* __restrict__ hid_in,
          const int* __restrict__ chars, float* __restrict__ x) {
  __shared__ float hid_s[H_];
  __shared__ float hproj_s[H_];
  __shared__ float sv_s[H_];
  __shared__ float e_s[T_];
  __shared__ float red_s[512];

  int tid = threadIdx.x;
  int b = blockIdx.x;
  int lane = tid & 31;
  int wave = tid >> 5;

  hid_s[tid] = hid_in[b * H_ + tid];
  sv_s[tid]  = sv[tid];
  __syncthreads();

  // hproj[h] = h2h_b[h] + hidden . h2h_w[h,:]
  {
    const v8h* wrow = (const v8h*)(h2h16 + (size_t)tid * H_);
    float acc = h2h_b[tid];
#pragma unroll 8
    for (int ch = 0; ch < 64; ++ch) {
      v8h w = wrow[ch];
#pragma unroll
      for (int e = 0; e < 8; ++e) acc += (float)w[e] * hid_s[ch * 8 + e];
    }
    hproj_s[tid] = acc;
  }
  __syncthreads();

  // e[t] = sum_h sv[h] * tanh(fp[b,t,h] + hproj[h]); wave handles 49 t's.
  // Each lane owns h = lane*16 .. lane*16+15: cache hproj/sv in registers once.
  {
    float hp[16], sw[16];
#pragma unroll
    for (int e = 0; e < 16; ++e) {
      int h = lane * 16 + e;
      hp[e] = hproj_s[h];
      sw[e] = sv_s[h];
    }
    const _Float16* rowbase = fproj + ((size_t)b * T_ + (size_t)wave * 49) * H_;
    for (int i = 0; i < 49; ++i) {
      const _Float16* row = rowbase + (size_t)i * H_;
      if (i < 48) __builtin_prefetch((const void*)(row + H_ + lane * 16), 0, 0);
      v16h v = *(const v16h*)(row + lane * 16);
      float partial = 0.0f;
#pragma unroll
      for (int e = 0; e < 16; ++e)
        partial += sw[e] * fast_tanh((float)v[e] + hp[e]);
#pragma unroll
      for (int off = 16; off > 0; off >>= 1) partial += __shfl_xor(partial, off, 32);
      if (lane == 0) e_s[wave * 49 + i] = partial;
    }
  }
  __syncthreads();

  // softmax over T
  float lm = -1e30f;
  for (int t = tid; t < T_; t += 512) lm = fmaxf(lm, e_s[t]);
  red_s[tid] = lm;
  __syncthreads();
  for (int s = 256; s > 0; s >>= 1) {
    if (tid < s) red_s[tid] = fmaxf(red_s[tid], red_s[tid + s]);
    __syncthreads();
  }
  float mx = red_s[0];
  __syncthreads();
  float ls = 0.0f;
  for (int t = tid; t < T_; t += 512) { float v = __expf(e_s[t] - mx); e_s[t] = v; ls += v; }
  red_s[tid] = ls;
  __syncthreads();
  for (int s = 256; s > 0; s >>= 1) {
    if (tid < s) red_s[tid] += red_s[tid + s];
    __syncthreads();
  }
  float inv = 1.0f / red_s[0];

  // context[c] = sum_t alpha[t] * featT[b,t,c]; write x = [context | onehot | pad]
  {
    const _Float16* fb = featT + (size_t)b * T_ * C_ + tid;
    float acc = 0.0f;
#pragma unroll 4
    for (int t = 0; t < T_; ++t) acc += e_s[t] * (float)fb[(size_t)t * C_];
    x[b * XK + tid] = acc * inv;
  }
  if (tid < 32) {
    int k = tid;
    float v = (k < NC_ && k == chars[b]) ? 1.0f : 0.0f;
    x[b * XK + C_ + k] = v;
  }
}

// ---------------- Per-step kernel 2: GRU cell (WMMA, LDS-staged weights) ----------------
// 32 blocks (one per 16-wide H slice), 128 threads (wave = M tile of 16 rows).
// The 6 weight tile-columns for this slice (3x17 ih + 3x16 hh = 99 tiles, ~99KB)
// are staged once into LDS (CDNA5: up to 320KB/WG) so all 4 waves share them.
__global__ void __launch_bounds__(128)
gru_step(const float* __restrict__ x, const float* __restrict__ hid_in,
         const _Float16* __restrict__ pkWih, const _Float16* __restrict__ pkWhh,
         const float* __restrict__ b_ih, const float* __restrict__ b_hh,
         float* __restrict__ hid_out) {
  extern __shared__ _Float16 bsh[];   // 99 * 512 halves
  int hs = blockIdx.x;           // 0..31
  int tid = threadIdx.x;
  int lane = tid & 31;
  int mt = tid >> 5;             // 0..3

  // cooperative stage of B tiles: tiles 0..50 = ih gates, 51..98 = hh gates
  for (int i = tid; i < 99 * 32; i += 128) {
    int l2 = i & 31;
    int tile = i >> 5;
    const _Float16* src;
    if (tile < 51) {
      int gate = tile / 17, kt = tile % 17;
      src = pkWih + (((size_t)(gate * 32 + hs) * 17 + kt) * 32 + l2) * 16;
    } else {
      int t2 = tile - 51;
      int gate = t2 / 16, kt = t2 % 16;
      src = pkWhh + (((size_t)(gate * 32 + hs) * 16 + kt) * 32 + l2) * 16;
    }
    *(v16h*)(bsh + ((size_t)tile * 32 + l2) * 16) = *(const v16h*)src;
  }
  __syncthreads();

  v8f air = {}, aiz = {}, ain = {}, ahr = {}, ahz = {}, ahn = {};
#pragma unroll 1
  for (int kt = 0; kt < 17; ++kt) {   // K = 544
    v16h a = load_a_f32(x, XK, mt * 16, kt, lane);
    air = wmma16(a, *(const v16h*)(bsh + ((0 * 17 + kt) * 32 + lane) * 16), air);
    aiz = wmma16(a, *(const v16h*)(bsh + ((1 * 17 + kt) * 32 + lane) * 16), aiz);
    ain = wmma16(a, *(const v16h*)(bsh + ((2 * 17 + kt) * 32 + lane) * 16), ain);
  }
#pragma unroll 1
  for (int kt = 0; kt < 16; ++kt) {   // K = 512
    v16h a = load_a_f32(hid_in, H_, mt * 16, kt, lane);
    ahr = wmma16(a, *(const v16h*)(bsh + ((51 + 0 * 16 + kt) * 32 + lane) * 16), ahr);
    ahz = wmma16(a, *(const v16h*)(bsh + ((51 + 1 * 16 + kt) * 32 + lane) * 16), ahz);
    ahn = wmma16(a, *(const v16h*)(bsh + ((51 + 2 * 16 + kt) * 32 + lane) * 16), ahn);
  }

  int n = hs * 16 + (lane & 15);
  int m0 = mt * 16 + (lane >> 4) * 8;
  float bir = b_ih[n], biz = b_ih[H_ + n], bin = b_ih[2 * H_ + n];
  float bhr = b_hh[n], bhz = b_hh[H_ + n], bhn = b_hh[2 * H_ + n];
#pragma unroll
  for (int r = 0; r < 8; ++r) {
    int m = m0 + r;
    float rg = sigm(air[r] + bir + ahr[r] + bhr);
    float zg = sigm(aiz[r] + biz + ahz[r] + bhz);
    float ng = fast_tanh(ain[r] + bin + rg * (ahn[r] + bhn));
    float hp = hid_in[(size_t)m * H_ + n];
    hid_out[(size_t)m * H_ + n] = (1.0f - zg) * ng + zg * hp;
  }
}

// ---------------- Per-step kernel 3: head mid layers (sg1 / lg1) ----------------
// 64 blocks: [0..31] -> sg1 n-tiles, [32..63] -> lg1 n-tiles; 128 threads
__global__ void __launch_bounds__(128)
heads_mid(const float* __restrict__ hid, const _Float16* __restrict__ pkSg1,
          const float* __restrict__ sg1_b, const _Float16* __restrict__ pkLg1,
          const float* __restrict__ lg1_b, float* __restrict__ s1,
          float* __restrict__ l1) {
  int sel = blockIdx.x >> 5;
  int nt  = blockIdx.x & 31;
  int lane = threadIdx.x & 31;
  int mt = threadIdx.x >> 5;
  const _Float16* pk = sel ? pkLg1 : pkSg1;
  const float* bias  = sel ? lg1_b : sg1_b;
  float* out         = sel ? l1 : s1;

  v8f acc = {};
#pragma unroll 2
  for (int kt = 0; kt < 16; ++kt) {
    v16h a = load_a_f32(hid, H_, mt * 16, kt, lane);
    acc = wmma16(a, load_b_tile(pk, 16, nt, kt, lane), acc);
  }
  int n = nt * 16 + (lane & 15);
  int m0 = mt * 16 + (lane >> 4) * 8;
  float bv = bias[n];
#pragma unroll
  for (int r = 0; r < 8; ++r) out[(size_t)(m0 + r) * H_ + n] = acc[r] + bv;
}

// ---------------- Per-step kernel 4: logits/loc + softmax + argmax ----------------
// one block, 128 threads (wave = M tile)
__global__ void __launch_bounds__(128)
heads_final(const float* __restrict__ s1, const float* __restrict__ l1,
            const _Float16* __restrict__ pkSg2, const float* __restrict__ sg2_b,
            const _Float16* __restrict__ pkLg2, const float* __restrict__ lg2_b,
            float* __restrict__ probs_out, float* __restrict__ loc_out,
            int* __restrict__ chars, int step) {
  __shared__ float lg[B_ * 32];
  int tid = threadIdx.x;
  int lane = tid & 31;
  int mt = tid >> 5;

  v8f a0 = {}, a1 = {}, a2 = {};
#pragma unroll 2
  for (int kt = 0; kt < 16; ++kt) {
    v16h a = load_a_f32(s1, H_, mt * 16, kt, lane);
    a0 = wmma16(a, load_b_tile(pkSg2, 16, 0, kt, lane), a0);
    a1 = wmma16(a, load_b_tile(pkSg2, 16, 1, kt, lane), a1);
    v16h al = load_a_f32(l1, H_, mt * 16, kt, lane);
    a2 = wmma16(al, load_b_tile(pkLg2, 16, 0, kt, lane), a2);
  }
  int n = lane & 15;
  int m0 = mt * 16 + (lane >> 4) * 8;
  float b0 = (n < NC_) ? sg2_b[n] : 0.0f;
  float b1 = (16 + n < NC_) ? sg2_b[16 + n] : 0.0f;
#pragma unroll
  for (int r = 0; r < 8; ++r) {
    lg[(m0 + r) * 32 + n]      = a0[r] + b0;
    lg[(m0 + r) * 32 + 16 + n] = a1[r] + b1;
  }
  if (n < LOC_) {
    float bl = lg2_b[n];
#pragma unroll
    for (int r = 0; r < 8; ++r) {
      int b = m0 + r;
      loc_out[(size_t)b * STEPS * LOC_ + (size_t)step * LOC_ + n] = sigm(a2[r] + bl);
    }
  }
  __syncthreads();

  if (tid < B_) {
    int b = tid;
    const float* row = lg + b * 32;
    float mx = row[0];
    int am = 0;
#pragma unroll
    for (int k = 1; k < NC_; ++k) {
      float v = row[k];
      if (v > mx) { mx = v; am = k; }
    }
    float sum = 0.0f;
    float ex[NC_];
#pragma unroll
    for (int k = 0; k < NC_; ++k) { ex[k] = __expf(row[k] - mx); sum += ex[k]; }
    float inv = 1.0f / sum;
    float* pr = probs_out + (size_t)b * STEPS * NC_ + (size_t)step * NC_;
#pragma unroll
    for (int k = 0; k < NC_; ++k) pr[k] = ex[k] * inv;
    chars[b] = am;
  }
}

// ---------------- host ----------------

extern "C" void kernel_launch(void* const* d_in, const int* in_sizes, int n_in,
                              void* d_out, int out_size, void* d_ws, size_t ws_size,
                              hipStream_t stream) {
  const float* fea      = (const float*)d_in[0];
  const float* i2h_w    = (const float*)d_in[1];
  const float* h2h_w    = (const float*)d_in[2];
  const float* h2h_b    = (const float*)d_in[3];
  const float* score_w  = (const float*)d_in[4];
  const float* gru_w_ih = (const float*)d_in[5];
  const float* gru_w_hh = (const float*)d_in[6];
  const float* gru_b_ih = (const float*)d_in[7];
  const float* gru_b_hh = (const float*)d_in[8];
  const float* sg1_w    = (const float*)d_in[9];
  const float* sg1_b    = (const float*)d_in[10];
  const float* sg2_w    = (const float*)d_in[11];
  const float* sg2_b    = (const float*)d_in[12];
  const float* lg1_w    = (const float*)d_in[13];
  const float* lg1_b    = (const float*)d_in[14];
  const float* lg2_w    = (const float*)d_in[15];
  const float* lg2_b    = (const float*)d_in[16];

  char* ws = (char*)d_ws;
  size_t off = 0;
  auto alloc = [&](size_t bytes) -> void* {
    void* p = ws + off;
    off = (off + bytes + 255) & ~(size_t)255;
    return p;
  };

  _Float16* featT  = (_Float16*)alloc((size_t)B_ * T_ * C_ * 2);
  _Float16* fproj  = (_Float16*)alloc((size_t)B_ * T_ * H_ * 2);
  _Float16* pkI2h  = (_Float16*)alloc((size_t)H_ * C_ * 2);
  _Float16* h2h16  = (_Float16*)alloc((size_t)H_ * H_ * 2);
  _Float16* pkWih  = (_Float16*)alloc((size_t)GN * XK * 2);
  _Float16* pkWhh  = (_Float16*)alloc((size_t)GN * H_ * 2);
  _Float16* pkSg1  = (_Float16*)alloc((size_t)H_ * H_ * 2);
  _Float16* pkSg2  = (_Float16*)alloc((size_t)32 * H_ * 2);
  _Float16* pkLg1  = (_Float16*)alloc((size_t)H_ * H_ * 2);
  _Float16* pkLg2  = (_Float16*)alloc((size_t)16 * H_ * 2);
  float* hidA      = (float*)alloc((size_t)B_ * H_ * 4);
  float* hidB      = (float*)alloc((size_t)B_ * H_ * 4);
  float* xbuf      = (float*)alloc((size_t)B_ * XK * 4);
  float* s1buf     = (float*)alloc((size_t)B_ * H_ * 4);
  float* l1buf     = (float*)alloc((size_t)B_ * H_ * 4);
  int*   chars     = (int*)alloc((size_t)B_ * 4);

  float* probs_out = (float*)d_out;
  float* loc_out   = (float*)d_out + (size_t)B_ * STEPS * NC_;

  // phase 0
  {
    size_t n = (size_t)B_ * T_ * C_;
    transpose_feat<<<(unsigned)((n + 255) / 256), 256, 0, stream>>>(fea, featT);
  }
  auto pack = [&](const float* W, _Float16* out, int N, int K, int Np, int Kp) {
    int total = (Np / 16) * (Kp / 32) * 512;
    pack_weight<<<(total + 255) / 256, 256, 0, stream>>>(W, out, N, K, Np, Kp);
  };
  pack(i2h_w, pkI2h, H_, C_, H_, C_);
  pack(gru_w_ih, pkWih, GN, 542, GN, XK);
  pack(gru_w_hh, pkWhh, GN, H_, GN, H_);
  pack(sg1_w, pkSg1, H_, H_, H_, H_);
  pack(sg2_w, pkSg2, NC_, H_, 32, H_);
  pack(lg1_w, pkLg1, H_, H_, H_, H_);
  pack(lg2_w, pkLg2, LOC_, H_, 16, H_);
  cvt_f32_f16<<<(H_ * H_ + 255) / 256, 256, 0, stream>>>(h2h_w, h2h16, H_ * H_);
  init_state<<<(B_ * H_ + 255) / 256, 256, 0, stream>>>(hidA, chars);
  featproj_gemm<<<(B_ * 49 * 32) / 8, 256, 0, stream>>>(featT, pkI2h, fproj);

  // sequential decode
  const unsigned gruLds = 99u * 512u * 2u;   // 99 B-tiles staged in LDS (~99KB)
  float* hin = hidA;
  float* hout = hidB;
  for (int s = 0; s < STEPS; ++s) {
    attn_step<<<B_, 512, 0, stream>>>(featT, fproj, h2h16, h2h_b, score_w,
                                      hin, chars, xbuf);
    gru_step<<<32, 128, gruLds, stream>>>(xbuf, hin, pkWih, pkWhh,
                                          gru_b_ih, gru_b_hh, hout);
    heads_mid<<<64, 128, 0, stream>>>(hout, pkSg1, sg1_b, pkLg1, lg1_b, s1buf, l1buf);
    heads_final<<<1, 128, 0, stream>>>(s1buf, l1buf, pkSg2, sg2_b, pkLg2, lg2_b,
                                       probs_out, loc_out, chars, s);
    float* tmp = hin; hin = hout; hout = tmp;
  }
  (void)in_sizes; (void)n_in; (void)out_size; (void)ws_size;
}
